// GnnLayer_70523363000699
// MI455X (gfx1250) — matrix operand order repaired and checked by reading, not compiled, not run on hardware
//
#include <hip/hip_runtime.h>
#include <hip/hip_bf16.h>

typedef __attribute__((ext_vector_type(16))) _Float16 v16h;
typedef __attribute__((ext_vector_type(8)))  _Float16 v8h;
typedef __attribute__((ext_vector_type(8)))  float    v8f;

#define KN    32
#define FIN   64
#define FOUT  64
#define RDIM  (KN * FIN)        // 2048 reduction dim
#define MBLK  16                // nodes per workgroup
#define NSTEPS (RDIM / 32)      // 64 WMMA k-steps
#define WOW_STRIDE (RDIM + 8)   // halfs; +8 halfs (16B) pad to spread LDS banks

// ---------------------------------------------------------------------------
// Pack weight [2048 x 64] f32 -> f16 fragments in exact WMMA B-operand layout.
// Fragment (kk, wt): lane L (g=L>>4, n=L&15) holds element e as
//   weight[(kk*32 + 8*g + e + (e>=8 ? 8 : 0)) * 64 + wt*16 + n]
// so each WMMA wave loads one contiguous v16h (32B) per lane per k-step.
// ---------------------------------------------------------------------------
__global__ void gnn_pack_weight(const float* __restrict__ weight,
                                _Float16* __restrict__ pb) {
    int i = blockIdx.x * blockDim.x + threadIdx.x;
    if (i >= NSTEPS * 4 * 32 * 16) return;
    int e    = i & 15;
    int lane = (i >> 4) & 31;
    int wt   = (i >> 9) & 3;
    int kk   = i >> 11;
    int g = lane >> 4;
    int n = lane & 15;
    int r = kk * 32 + 8 * g + e + ((e >> 3) << 3);
    pb[i] = (_Float16)weight[r * FOUT + wt * 16 + n];
}

// ---------------------------------------------------------------------------
// Main kernel: 128 threads (4 wave32) per block, 16 nodes per block.
//   wave w computes output columns [w*16, w*16+16) for all 16 nodes.
// ---------------------------------------------------------------------------
__global__ __launch_bounds__(128) void gnn_wmma_kernel(
    const float* __restrict__ h,
    const float* __restrict__ pos,
    const int*   __restrict__ nidx,
    const _Float16* __restrict__ pb,     // packed B fragments (d_ws)
    const float* __restrict__ bias,
    float*       __restrict__ out,
    int n_nodes) {

    extern __shared__ char smem[];
    int*      nbr_s  = (int*)smem;                 // 512 ints   (2 KB)
    float*    invd_s = (float*)(smem + 2048);      // 512 floats (2 KB)
    _Float16* wow    = (_Float16*)(smem + 4096);   // 16 x WOW_STRIDE halfs

    const int tid  = threadIdx.x;
    const int base = blockIdx.x * MBLK;

    // ---- phase 0: neighbor ids + inverse distances into LDS ----
    for (int j = tid; j < MBLK * KN; j += 128) {
        int m = j >> 5;
        int k = j & 31;
        int node = base + m;
        if (node >= n_nodes) node = n_nodes - 1;   // clamp (stores guarded)
        int nb = nidx[node * KN + k];
        nbr_s[j] = nb;
        float dx = pos[node * 3 + 0] - pos[nb * 3 + 0];
        float dy = pos[node * 3 + 1] - pos[nb * 3 + 1];
        float dz = pos[node * 3 + 2] - pos[nb * 3 + 2];
        float sq = dx * dx + dy * dy + dz * dz;
        // reference: dist = 0.5 when sq==0 else sqrt(sq); we need 1/dist
        invd_s[j] = (sq == 0.0f) ? 2.0f : (1.0f / sqrtf(sq));
    }
    __syncthreads();

    // ---- phase 1: gather h[nbr] * invd -> f16 wow tile in LDS ----
    // consecutive tids read consecutive features -> coalesced 256B rows
    for (int i = tid; i < MBLK * RDIM; i += 128) {
        int m = i >> 11;            // /RDIM
        int r = i & (RDIM - 1);
        int k = r >> 6;
        int f = r & 63;
        float v = h[(size_t)nbr_s[m * KN + k] * FIN + f] * invd_s[m * KN + k];
        wow[m * WOW_STRIDE + r] = (_Float16)v;
    }
    __syncthreads();

    // ---- phase 2: 64 x v_wmma_f32_16x16x32_f16 per wave ----
    const int lane = tid & 31;
    const int wv   = tid >> 5;      // n-tile 0..3
    const int g    = lane >> 4;     // half-wave group
    const int ln15 = lane & 15;     // A row / C column

    v8f acc = {};

    // A fragment: lane (g, m=ln15) holds K = kk*32 + 8g + e (e<8)
    //                        and K = kk*32 + 16 + 8g + (e-8) (e>=8)
    const _Float16* arow = wow + ln15 * WOW_STRIDE + 8 * g;
    // B fragment: contiguous v16h per lane from packed weights
    const v16h* pbw = ((const v16h*)pb) + (size_t)wv * 32 + lane;

    union AFrag { v16h v; v8h h2[2]; };

    #pragma unroll 4
    for (int kk = 0; kk < NSTEPS; ++kk) {
        AFrag a;
        a.h2[0] = *(const v8h*)(arow + kk * 32);        // ds_load_b128
        a.h2[1] = *(const v8h*)(arow + kk * 32 + 16);   // ds_load_b128
        v16h b = pbw[(size_t)kk * 4 * 32];              // 32B coalesced global
        acc = __builtin_amdgcn_wmma_f32_16x16x32_f16(
            /*neg_a=*/false, a.v, /*neg_b=*/false, b,
            /*c_mod=*/(short)0, acc, /*reuse_a=*/false, /*reuse_b=*/false);
    }

    // ---- epilogue: bias + leaky_relu(0.01) + store ----
    // C layout: vgpr p, lane -> row = p + 8*g, col = ln15
    const float bv = bias[wv * 16 + ln15];
    float* outp = out + (size_t)(base + 8 * g) * FOUT + wv * 16 + ln15;

    if (base + MBLK <= n_nodes) {
        // fast path: full block, unguarded coalesced stores
        #pragma unroll
        for (int p = 0; p < 8; ++p) {
            float x = acc[p] + bv;
            x = (x > 0.0f) ? x : 0.01f * x;
            outp[(size_t)p * FOUT] = x;
        }
    } else {
        // tail block: per-row guard
        #pragma unroll
        for (int p = 0; p < 8; ++p) {
            if (base + p + 8 * g < n_nodes) {
                float x = acc[p] + bv;
                x = (x > 0.0f) ? x : 0.01f * x;
                outp[(size_t)p * FOUT] = x;
            }
        }
    }
}

extern "C" void kernel_launch(void* const* d_in, const int* in_sizes, int n_in,
                              void* d_out, int out_size, void* d_ws, size_t ws_size,
                              hipStream_t stream) {
    const float*    h      = (const float*)d_in[0];
    const float*    pos    = (const float*)d_in[1];
    const int*      nidx   = (const int*)d_in[2];
    const float*    weight = (const float*)d_in[3];
    const float*    bias   = (const float*)d_in[4];
    float*          out    = (float*)d_out;
    _Float16*       pb     = (_Float16*)d_ws;     // 256 KB packed B fragments

    int n_nodes = in_sizes[0] / FIN;

    // 1) pack weight into WMMA B-fragment layout (f32 -> f16)
    int pack_elems = NSTEPS * 4 * 32 * 16;        // 131072
    gnn_pack_weight<<<(pack_elems + 255) / 256, 256, 0, stream>>>(weight, pb);

    // 2) fused gather + scale + WMMA GEMM + bias + leaky-ReLU
    int nblocks = (n_nodes + MBLK - 1) / MBLK;    // 3125 for N=50000
    size_t smem_bytes = 4096 + (size_t)MBLK * WOW_STRIDE * sizeof(_Float16); // 69888
    gnn_wmma_kernel<<<nblocks, 128, smem_bytes, stream>>>(
        h, pos, nidx, pb, bias, out, n_nodes);
}
